// AttentionConv_33019708572307
// MI455X (gfx1250) — matrix operand order repaired and checked by reading, not compiled, not run on hardware
//
#include <hip/hip_runtime.h>

// ---------------------------------------------------------------------------
// Fused AttentionConv for MI455X (gfx1250, wave32, WMMA).
// GEMMs use split-bf16 (hi/lo) emulation of f32 on V_WMMA_F32_16X16X32_BF16:
//   W*X ~= Wh*Xh + Wh*Xl + Wl*Xh   (f32 accumulation, ~f32 accuracy)
// Kernel 1 (prep): convert+swizzle weights into A-fragment layout in d_ws.
// Kernel 2 (fused): stage x (hi/lo, transposed) in LDS, WMMA Q/K/V, fused
// 7-tap softmax-attention epilogue, coalesced f32 store.
// ---------------------------------------------------------------------------

typedef __attribute__((ext_vector_type(8)))  float  v8f;
typedef __attribute__((ext_vector_type(8)))  __bf16 v8bf;
typedef __attribute__((ext_vector_type(16))) __bf16 v16bf;

#define C_IN   256
#define C_OUT  256
#define W_DIM  4096
#define TILE_W 32
#define HALO   8                    // halo per side (>=3, rounded to n-tiles)
#define COLS   (TILE_W + 2*HALO)    // 48 staged columns
#define KB32   (C_IN / 32)          // 8 wmma K-steps of 32
#define MTILES (C_OUT / 16)         // 16

#define XSTR   264                  // xsT row stride in bf16 (256 + 8 pad)
#define FSTR   50                   // k/v f32 LDS row stride (48 + 2 pad)

// per (matrix, hi/lo): [mtile][kb][half][lane][8] bf16 = 65536 elements
#define WFRAG_ELEMS (MTILES * KB32 * 2 * 32 * 8)
#define PREP_THREADS (3 * MTILES * KB32 * 2 * 32)   // 24576

// ---- bf16 helpers ---------------------------------------------------------
__device__ __forceinline__ unsigned short f32_to_bf16_rne(float f) {
  unsigned int u = __float_as_uint(f);
  unsigned int r = u + 0x7FFFu + ((u >> 16) & 1u);
  return (unsigned short)(r >> 16);
}
__device__ __forceinline__ float bf16_bits_to_f32(unsigned short h) {
  return __uint_as_float(((unsigned int)h) << 16);
}
__device__ __forceinline__ v16bf cat8(v8bf a, v8bf b) {
  return __builtin_shufflevector(a, b, 0, 1, 2, 3, 4, 5, 6, 7,
                                 8, 9, 10, 11, 12, 13, 14, 15);
}
__device__ __forceinline__ v8f wmma_bf16(v16bf a, v16bf b, v8f c) {
  return __builtin_amdgcn_wmma_f32_16x16x32_bf16(
      /*neg_a=*/false, a, /*neg_b=*/false, b,
      /*c_mod=*/(short)0, c, /*reuse_a=*/false, /*reuse_b=*/false);
}

// ---- Kernel 1: weight conversion + A-fragment swizzle into d_ws ----------
// A 16x32 bf16 fragment layout (wave32): lane m = lane&15;
//   lanes 0-15:  VGPR0-3 = K {0..7},   VGPR4-7 = K {16..23}
//   lanes 16-31: VGPR0-3 = K {8..15},  VGPR4-7 = K {24..31}
// ws layout per (mat,comp): idx = (((mt*KB32 + kb)*2 + half)*32 + lane)*8
__global__ void prep_weights(const float* __restrict__ wq,
                             const float* __restrict__ wk,
                             const float* __restrict__ wv,
                             unsigned short* __restrict__ wsA) {
  int t = blockIdx.x * blockDim.x + threadIdx.x;
  if (t >= PREP_THREADS) return;
  int lane = t & 31;
  int half = (t >> 5) & 1;
  int kb   = (t >> 6) & (KB32 - 1);
  int mt   = (t >> 9) & (MTILES - 1);
  int mat  = t >> 13;   // 0=q, 1=k, 2=v

  const float* Wt = (mat == 0) ? wq : (mat == 1) ? wk : wv;
  int row   = mt * 16 + (lane & 15);
  int kbase = kb * 32 + ((lane >> 4) ? 8 : 0) + (half ? 16 : 0);
  const float* src = Wt + row * C_IN + kbase;

  unsigned short* dhi = wsA + ((size_t)mat * 2 + 0) * WFRAG_ELEMS +
                        ((size_t)(t & 8191)) * 8;
  unsigned short* dlo = wsA + ((size_t)mat * 2 + 1) * WFRAG_ELEMS +
                        ((size_t)(t & 8191)) * 8;
#pragma unroll
  for (int i = 0; i < 8; ++i) {
    float f = src[i];
    unsigned short h = f32_to_bf16_rne(f);
    float residual = f - bf16_bits_to_f32(h);
    dhi[i] = h;
    dlo[i] = f32_to_bf16_rne(residual);
  }
}

// ---- fragment loaders -----------------------------------------------------
__device__ __forceinline__ v16bf load_afrag(const unsigned short* wsbase,
                                            int mt, int kb, int lane) {
  const unsigned short* p = wsbase + ((size_t)((mt * KB32 + kb) * 2) * 32 + lane) * 8;
  v8bf c1 = *(const v8bf*)p;             // half 0 -> VGPR0-3
  v8bf c2 = *(const v8bf*)(p + 32 * 8);  // half 1 -> VGPR4-7
  return cat8(c1, c2);
}
// B 32x16 bf16 fragment: lane n = lane&15; lanes 0-15 K {0..7,16..23},
// lanes 16-31 K {8..15,24..31}; xsT is [col][ch] so this is two 16B runs per lane.
__device__ __forceinline__ v16bf load_bfrag(const unsigned short* xsT,
                                            int kb, int colbase, int lane) {
  int col = colbase + (lane & 15);
  int ch0 = kb * 32 + ((lane >> 4) << 3);
  const unsigned short* p = xsT + col * XSTR + ch0;
  v8bf c1 = *(const v8bf*)p;
  v8bf c2 = *(const v8bf*)(p + 16);
  return cat8(c1, c2);
}
// D tile (16x16 f32) -> LDS: VGPR r holds row m0+r (lanes 0-15) / m0+r+8
__device__ __forceinline__ void store_dtile(float* dst, const v8f& acc,
                                            int m0, int ncol0, int lhalf, int khalf) {
#pragma unroll
  for (int r = 0; r < 8; ++r)
    dst[(m0 + r + 8 * khalf) * FSTR + (ncol0 + lhalf)] = acc[r];
}

// ---- Kernel 2: fused Q/K/V GEMM + windowed softmax attention -------------
__global__ void __launch_bounds__(256, 2)
attn_conv_fused(const float* __restrict__ x,
                const unsigned short* __restrict__ wsA,
                float* __restrict__ out) {
  extern __shared__ unsigned char smem_raw[];
  unsigned short* xsh = (unsigned short*)smem_raw;          // 48 x XSTR bf16 hi
  unsigned short* xsl = xsh + COLS * XSTR;                  // 48 x XSTR bf16 lo
  float* ks = (float*)(xsl + COLS * XSTR);                  // 256 x FSTR f32
  float* vs = ks + C_OUT * FSTR;                            // 256 x FSTR f32

  const int tid   = threadIdx.x;
  const int wave  = tid >> 5;
  const int lane  = tid & 31;
  const int lhalf = lane & 15;
  const int khalf = lane >> 4;

  const int b  = blockIdx.y;
  const int w0 = blockIdx.x * TILE_W;

  const unsigned short* wsQh = wsA + 0 * (size_t)WFRAG_ELEMS;
  const unsigned short* wsQl = wsA + 1 * (size_t)WFRAG_ELEMS;
  const unsigned short* wsKh = wsA + 2 * (size_t)WFRAG_ELEMS;
  const unsigned short* wsKl = wsA + 3 * (size_t)WFRAG_ELEMS;
  const unsigned short* wsVh = wsA + 4 * (size_t)WFRAG_ELEMS;
  const unsigned short* wsVl = wsA + 5 * (size_t)WFRAG_ELEMS;

  // ---- Stage x tile (+halo) as transposed hi/lo bf16, zero-padded --------
  const float* xb = x + (size_t)b * C_IN * W_DIM;
  for (int idx = tid; idx < C_IN * COLS; idx += 256) {
    int c  = idx % COLS;   // consecutive tids -> consecutive w (coalesced read)
    int ch = idx / COLS;
    int w  = w0 - HALO + c;
    float f = (w >= 0 && w < W_DIM) ? xb[ch * W_DIM + w] : 0.0f;
    unsigned short h = f32_to_bf16_rne(f);
    xsh[c * XSTR + ch] = h;
    xsl[c * XSTR + ch] = f32_to_bf16_rne(f - bf16_bits_to_f32(h));
  }
  __syncthreads();

  // ---- Per wave: 2 M-tiles (16 output channels each) ---------------------
#pragma unroll
  for (int mi = 0; mi < 2; ++mi) {
    const int mt = wave * 2 + mi;
    const int m0 = mt * 16;

    // ---- K and V over all 48 staged columns (3 n-tiles) ----
    v8f aK0 = {}, aK1 = {}, aK2 = {};
    v8f aV0 = {}, aV1 = {}, aV2 = {};
    for (int kb = 0; kb < KB32; ++kb) {
      v16bf akh = load_afrag(wsKh, mt, kb, lane);
      v16bf akl = load_afrag(wsKl, mt, kb, lane);
      v16bf avh = load_afrag(wsVh, mt, kb, lane);
      v16bf avl = load_afrag(wsVl, mt, kb, lane);
#pragma unroll
      for (int n = 0; n < 3; ++n) {
        v16bf bh = load_bfrag(xsh, kb, n * 16, lane);
        v16bf bl = load_bfrag(xsl, kb, n * 16, lane);
        v8f* accK = (n == 0) ? &aK0 : (n == 1) ? &aK1 : &aK2;
        v8f* accV = (n == 0) ? &aV0 : (n == 1) ? &aV1 : &aV2;
        *accK = wmma_bf16(akh, bh, *accK);
        *accK = wmma_bf16(akh, bl, *accK);
        *accK = wmma_bf16(akl, bh, *accK);
        *accV = wmma_bf16(avh, bh, *accV);
        *accV = wmma_bf16(avh, bl, *accV);
        *accV = wmma_bf16(avl, bh, *accV);
      }
    }
    store_dtile(ks, aK0, m0, 0,  lhalf, khalf);
    store_dtile(ks, aK1, m0, 16, lhalf, khalf);
    store_dtile(ks, aK2, m0, 32, lhalf, khalf);
    store_dtile(vs, aV0, m0, 0,  lhalf, khalf);
    store_dtile(vs, aV1, m0, 16, lhalf, khalf);
    store_dtile(vs, aV2, m0, 32, lhalf, khalf);

    // ---- Q over the central 32 columns (2 n-tiles), kept in registers ----
    v8f aQ0 = {}, aQ1 = {};
    for (int kb = 0; kb < KB32; ++kb) {
      v16bf aqh = load_afrag(wsQh, mt, kb, lane);
      v16bf aql = load_afrag(wsQl, mt, kb, lane);
#pragma unroll
      for (int n = 0; n < 2; ++n) {
        v16bf bh = load_bfrag(xsh, kb, HALO + n * 16, lane);
        v16bf bl = load_bfrag(xsl, kb, HALO + n * 16, lane);
        v8f* acc = (n == 0) ? &aQ0 : &aQ1;
        *acc = wmma_bf16(aqh, bh, *acc);
        *acc = wmma_bf16(aqh, bl, *acc);
        *acc = wmma_bf16(aql, bh, *acc);
      }
    }

    // ---- Fused 7-tap softmax-attention epilogue ----
    // This wave wrote ks/vs rows [m0, m0+16) itself; same-wave LDS ops are
    // in-order, so no extra barrier is needed before reading them back.
#pragma unroll
    for (int n = 0; n < 2; ++n) {
      const int col = HALO + n * 16 + lhalf;
#pragma unroll
      for (int r = 0; r < 8; ++r) {
        const int ch = m0 + r + 8 * khalf;
        const float q = (n == 0) ? aQ0[r] : aQ1[r];
        float s[7], vv[7];
        float mx = -3.402823466e+38f;
#pragma unroll
        for (int j = 0; j < 7; ++j) {
          float kk = ks[ch * FSTR + col - 3 + j];
          vv[j] = vs[ch * FSTR + col - 3 + j];
          s[j] = q * kk;
          mx = fmaxf(mx, s[j]);
        }
        float den = 0.0f, num = 0.0f;
#pragma unroll
        for (int j = 0; j < 7; ++j) {
          float e = __expf(s[j] - mx);
          den += e;
          num += e * vv[j];
        }
        out[((size_t)b * C_OUT + ch) * W_DIM + (w0 + n * 16 + lhalf)] = num / den;
      }
    }
  }
}

extern "C" void kernel_launch(void* const* d_in, const int* in_sizes, int n_in,
                              void* d_out, int out_size, void* d_ws, size_t ws_size,
                              hipStream_t stream) {
  (void)in_sizes; (void)n_in; (void)ws_size; (void)out_size;
  const float* x  = (const float*)d_in[0];
  const float* wq = (const float*)d_in[1];
  const float* wk = (const float*)d_in[2];
  const float* wv = (const float*)d_in[3];
  float* out = (float*)d_out;
  unsigned short* wsA = (unsigned short*)d_ws;  // needs 6*65536*2 B = 768 KB

  // Kernel 1: convert + swizzle weights (recomputed every call; deterministic)
  prep_weights<<<dim3((PREP_THREADS + 255) / 256), dim3(256), 0, stream>>>(
      wq, wk, wv, wsA);

  // Kernel 2: fused attention conv
  const size_t lds_bytes =
      (size_t)2 * COLS * XSTR * sizeof(unsigned short) +   // xs hi/lo: 50,688 B
      (size_t)2 * C_OUT * FSTR * sizeof(float);            // k,v:     102,400 B
  hipFuncSetAttribute((const void*)attn_conv_fused,
                      hipFuncAttributeMaxDynamicSharedMemorySize,
                      (int)lds_bytes);
  dim3 grid(W_DIM / TILE_W, 8);  // 128 x 8 = 1024 workgroups
  dim3 block(256);               // 8 waves (wave32)
  attn_conv_fused<<<grid, block, lds_bytes, stream>>>(x, wsA, out);
}
